// GAT_70446053589175
// MI455X (gfx1250) — matrix operand order repaired and checked by reading, not compiled, not run on hardware
//
#include <hip/hip_runtime.h>

// ---------------------------------------------------------------------------
// GAT forward for MI455X (gfx1250, wave32). Heavy math via
// v_wmma_f32_16x16x32_bf16 (fp32 accumulate). Two-pass masked softmax; the
// fused P@V pass builds the probability block once per j-step in the WMMA
// A-operand lane layout and reuses it across ALL output column tiles
// (NT accumulators per wave) so the v_exp_f32 stream is amortized and the
// loop is WMMA-dense. adj is compressed to a 2MB bitmask (L2-resident).
// ---------------------------------------------------------------------------

typedef __attribute__((ext_vector_type(16))) __bf16 v16bf;
typedef __attribute__((ext_vector_type(8)))  float  v8f;

struct __attribute__((aligned(16))) U4 { unsigned int x, y, z, w; };

union ABreg {
    v16bf          v;
    unsigned short u[16];
    U4             q[2];
};

__device__ __forceinline__ unsigned short f2bf(float f) {
    unsigned int u = __float_as_uint(f);
    u += 0x7FFFu + ((u >> 16) & 1u);          // round-to-nearest-even
    return (unsigned short)(u >> 16);
}

// --------------------------- small prep kernels ----------------------------

__global__ void cvt_f32_to_bf16(const float* __restrict__ in,
                                unsigned short* __restrict__ out, int total) {
    int i = blockIdx.x * blockDim.x + threadIdx.x;
    if (i < total) out[i] = f2bf(in[i]);
}

// W [h][f][d] -> Wt [h][d][f] (bf16)
__global__ void transpose_cvt(const float* __restrict__ W,
                              unsigned short* __restrict__ Wt,
                              int F, int D, int total) {
    int i = blockIdx.x * blockDim.x + threadIdx.x;
    if (i >= total) return;
    int per = F * D;
    int h = i / per;
    int r = i - h * per;
    int f = r / D;
    int d = r - f * D;
    Wt[(size_t)h * per + (size_t)d * F + f] = f2bf(W[i]);
}

// adj int32 [N,N] -> bitmask uint32 [N, N/32]
__global__ void pack_adj(const int* __restrict__ adj,
                         unsigned int* __restrict__ bits, int N, int words) {
    int w = blockIdx.x * blockDim.x + threadIdx.x;
    if (w >= words) return;
    int wpr = N >> 5;
    int row = w / wpr;
    int c0  = (w - row * wpr) << 5;
    const int* a = adj + (size_t)row * N + c0;
    unsigned int b = 0;
#pragma unroll
    for (int k = 0; k < 32; ++k) b |= (a[k] > 0 ? 1u : 0u) << k;
    bits[w] = b;
}

// ------------------------------ WMMA GEMM ----------------------------------
// Wave computes a 16 x (NT*16) strip: A tile loaded once per k-step, reused
// across NT B tiles / NT accumulators. C fp32 row-major + bf16 transposed.
template <int NT>
__global__ void gemm_bf16_tn(const unsigned short* __restrict__ A,
                             const unsigned short* __restrict__ Bt,
                             float* __restrict__ C,
                             unsigned short* __restrict__ CtB,
                             int M, int K, int n_heads) {
    constexpr int Nc = NT * 16;
    int wpb  = blockDim.x >> 5;
    int gw   = blockIdx.x * wpb + (threadIdx.x >> 5);
    int lane = threadIdx.x & 31;
    int strips = M >> 4;
    if (gw >= strips * n_heads) return;
    int h  = gw / strips;
    int m0 = (gw - h * strips) << 4;
    int half = lane >> 4, l15 = lane & 15;

    const unsigned short* Arow = A  + (size_t)(m0 + l15) * K;
    const unsigned short* Bcol = Bt + (size_t)h * Nc * K + (size_t)l15 * K;

    v8f acc[NT];
#pragma unroll
    for (int nt = 0; nt < NT; ++nt) acc[nt] = (v8f){};

    for (int kk = 0; kk < K; kk += 32) {
        ABreg a;
        // A 16x32 bf16 layout: lane<16 row holds K=0..7,16..23; lane>=16 holds 8..15,24..31
        a.q[0] = *(const U4*)(Arow + kk + half * 8);
        a.q[1] = *(const U4*)(Arow + kk + 16 + half * 8);
#pragma unroll
        for (int nt = 0; nt < NT; ++nt) {
            ABreg b;
            // B 32x16 bf16 layout: lanes0-15 col holds K=0..7,16..23; lanes16-31 hold 8..15,24..31
            const unsigned short* Bp = Bcol + (size_t)nt * 16 * K + kk;
            b.q[0] = *(const U4*)(Bp + half * 8);
            b.q[1] = *(const U4*)(Bp + 16 + half * 8);
            acc[nt] = __builtin_amdgcn_wmma_f32_16x16x32_bf16(false, a.v, false, b.v,
                                                              (short)0, acc[nt], false, false);
        }
    }
    float*          Ch  = C   + (size_t)h * M * Nc;
    unsigned short* Cth = CtB + (size_t)h * Nc * M;
#pragma unroll
    for (int nt = 0; nt < NT; ++nt) {
#pragma unroll
        for (int r = 0; r < 8; ++r) {
            int gm = m0 + r + half * 8;
            int gn = nt * 16 + l15;
            float v = acc[nt][r];
            Ch[(size_t)gm * Nc + gn] = v;
            Cth[(size_t)gn * M + gm] = f2bf(v);
        }
    }
}

// ----------------------- attention scalar projections ----------------------
__global__ void s_dots(const float* __restrict__ src,
                       const float* __restrict__ a1,
                       const float* __restrict__ a2,
                       float* __restrict__ s1, float* __restrict__ s2,
                       int Dd, int rows_per_head, int total_rows) {
    int r = blockIdx.x * blockDim.x + threadIdx.x;
    if (r >= total_rows) return;
    int h = r / rows_per_head;
    const float* row = src + (size_t)r * Dd;
    const float* v1  = a1 + (size_t)h * Dd;
    const float* v2  = a2 + (size_t)h * Dd;
    float acc1 = 0.f, acc2 = 0.f;
    for (int d = 0; d < Dd; ++d) {
        float w = row[d];
        acc1 = fmaf(w, v1[d], acc1);
        acc2 = fmaf(w, v2[d], acc2);
    }
    s1[r] = acc1;
    s2[r] = acc2;
}

// --------------------- pass 1: masked softmax statistics -------------------
// One wave per (head,row): m = max_j e_ij, Z = sum_j exp(e_ij - m), with
// e_ij = leakyrelu(s1_i + s2_j) where adj bit set, else -9e15 (matches ref).
__global__ void softmax_stats(const float* __restrict__ s1,
                              const float* __restrict__ s2,
                              const unsigned int* __restrict__ bits,
                              float* __restrict__ mrow, float* __restrict__ zrow,
                              int N, int rows_per_head, int total_rows) {
    int wave = (blockIdx.x * blockDim.x + threadIdx.x) >> 5;
    int lane = threadIdx.x & 31;
    if (wave >= total_rows) return;
    int h = wave / rows_per_head;
    int i = wave - h * rows_per_head;
    const float s1r = s1[wave];
    const float* s2h = s2 + (size_t)h * rows_per_head;
    const unsigned int* br = bits + (size_t)i * (N >> 5);

    float mx = -3.4e38f;
    for (int j = lane; j < N; j += 32) {
        float e = s1r + s2h[j];
        e = e > 0.f ? e : 0.2f * e;
        e = ((br[j >> 5] >> (j & 31)) & 1u) ? e : -9e15f;
        mx = fmaxf(mx, e);
    }
#pragma unroll
    for (int off = 16; off; off >>= 1) mx = fmaxf(mx, __shfl_xor(mx, off, 32));

    float sum = 0.f;
    for (int j = lane; j < N; j += 32) {
        float e = s1r + s2h[j];
        e = e > 0.f ? e : 0.2f * e;
        e = ((br[j >> 5] >> (j & 31)) & 1u) ? e : -9e15f;
        sum += __expf(e - mx);
    }
#pragma unroll
    for (int off = 16; off; off >>= 1) sum += __shfl_xor(sum, off, 32);

    if (lane == 0) { mrow[wave] = mx; zrow[wave] = sum; }
}

// --------------------- pass 2: fused softmax(P) @ V ------------------------
// Wave owns a 16-row x (NT*16)-col strip of head h. Streams j in steps of 32:
// builds the 16x32 probability block ONCE (exp amortized), then issues NT
// back-to-back WMMAs against the NT WhT column tiles.
template <int NT>
__global__ void attn_av(const float* __restrict__ s1,
                        const float* __restrict__ s2,
                        const float* __restrict__ mrow,
                        const float* __restrict__ zrow,
                        const unsigned int* __restrict__ bits,
                        const unsigned short* __restrict__ WhT,
                        int N,
                        float* __restrict__ outF,
                        unsigned short* __restrict__ outB,
                        int ldo, int elu) {
    constexpr int Dhead = NT * 16;
    int h    = blockIdx.y;
    int lane = threadIdx.x & 31;
    int w    = threadIdx.x >> 5;
    int i0   = (blockIdx.x * (blockDim.x >> 5) + w) << 4;
    int half = lane >> 4, l15 = lane & 15;

    size_t hb = (size_t)h * N;
    const float* s2h = s2 + hb;
    int   row = i0 + l15;
    float s1r = s1[hb + row];
    float mr  = mrow[hb + row];
    const unsigned int*   br   = bits + (size_t)row * (N >> 5);
    const unsigned short* Bcol = WhT + (size_t)h * Dhead * N + (size_t)l15 * N;

    v8f acc[NT];
#pragma unroll
    for (int nt = 0; nt < NT; ++nt) acc[nt] = (v8f){};

    for (int j0 = 0; j0 < N; j0 += 32) {
        unsigned int wbits = br[j0 >> 5];
        ABreg a;
#pragma unroll
        for (int s = 0; s < 16; ++s) {
            int k = ((s >> 3) << 4) + half * 8 + (s & 7);   // A-layout K index
            float e = s1r + s2h[j0 + k];
            e = e > 0.f ? e : 0.2f * e;
            e = ((wbits >> k) & 1u) ? e : -9e15f;
            a.u[s] = f2bf(__expf(e - mr));
        }
#pragma unroll
        for (int nt = 0; nt < NT; ++nt) {
            ABreg b;
            const unsigned short* Bp = Bcol + (size_t)nt * 16 * N + j0;
            b.q[0] = *(const U4*)(Bp + half * 8);
            b.q[1] = *(const U4*)(Bp + 16 + half * 8);
            acc[nt] = __builtin_amdgcn_wmma_f32_16x16x32_bf16(false, a.v, false, b.v,
                                                              (short)0, acc[nt], false, false);
        }
    }

    float zinv[8];
#pragma unroll
    for (int r = 0; r < 8; ++r) zinv[r] = 1.f / zrow[hb + i0 + r + half * 8];

#pragma unroll
    for (int nt = 0; nt < NT; ++nt) {
#pragma unroll
        for (int r = 0; r < 8; ++r) {
            int gr  = i0 + r + half * 8;
            float v = acc[nt][r] * zinv[r];
            if (elu) v = v > 0.f ? v : expm1f(v);
            int col = h * Dhead + nt * 16 + l15;
            if (outB) outB[(size_t)gr * ldo + col] = f2bf(v);
            else      outF[(size_t)gr * ldo + col] = v;
        }
    }
}

// ------------------------------- launcher ----------------------------------

extern "C" void kernel_launch(void* const* d_in, const int* in_sizes, int n_in,
                              void* d_out, int out_size, void* d_ws, size_t ws_size,
                              hipStream_t stream) {
    (void)in_sizes; (void)n_in; (void)out_size; (void)ws_size;
    constexpr int N = 4096, F = 512, D = 64, H = 8, O = 128;
    constexpr int HD = H * D;                  // 512

    const float* x      = (const float*)d_in[0];
    const int*   adj    = (const int*)d_in[1];
    const float* Wh_w   = (const float*)d_in[2];   // [H,F,D]
    const float* a1h    = (const float*)d_in[3];   // [H,D]
    const float* a2h    = (const float*)d_in[4];   // [H,D]
    const float* Wout   = (const float*)d_in[5];   // [HD,O]
    const float* a1o    = (const float*)d_in[6];   // [O]
    const float* a2o    = (const float*)d_in[7];   // [O]
    float*       out    = (float*)d_out;           // [N,O]

    // ---- workspace carve-up (256B aligned) ----
    char* ws = (char*)d_ws;
    size_t off = 0;
    auto carve = [&](size_t bytes) -> char* {
        char* p = ws + off;
        off += (bytes + 255) & ~(size_t)255;
        return p;
    };
    unsigned short* xb    = (unsigned short*)carve((size_t)N * F * 2);     // x bf16
    unsigned short* WtH   = (unsigned short*)carve((size_t)H * D * F * 2); // W_heads^T
    unsigned short* WtO   = (unsigned short*)carve((size_t)O * HD * 2);    // W_out^T
    unsigned int*   bits  = (unsigned int*)carve((size_t)N * (N / 32) * 4);
    float*          Whf   = (float*)carve((size_t)H * N * D * 4);          // Wh fp32
    unsigned short* WhT   = (unsigned short*)carve((size_t)H * D * N * 2);
    float*          s1h   = (float*)carve((size_t)H * N * 4);
    float*          s2h   = (float*)carve((size_t)H * N * 4);
    float*          mh    = (float*)carve((size_t)H * N * 4);
    float*          zh    = (float*)carve((size_t)H * N * 4);
    unsigned short* hcat  = (unsigned short*)carve((size_t)N * HD * 2);    // concat bf16
    float*          Wof   = (float*)carve((size_t)N * O * 4);              // Wh_out fp32
    unsigned short* WoT   = (unsigned short*)carve((size_t)O * N * 2);
    float*          s1o   = (float*)carve((size_t)N * 4);
    float*          s2o   = (float*)carve((size_t)N * 4);
    float*          mo    = (float*)carve((size_t)N * 4);
    float*          zo    = (float*)carve((size_t)N * 4);

    // 1) precision conversion / transposes / adj bitmask
    cvt_f32_to_bf16<<<(N * F + 255) / 256, 256, 0, stream>>>(x, xb, N * F);
    transpose_cvt<<<(H * F * D + 255) / 256, 256, 0, stream>>>(Wh_w, WtH, F, D, H * F * D);
    transpose_cvt<<<(HD * O + 255) / 256, 256, 0, stream>>>(Wout, WtO, HD, O, HD * O);
    pack_adj<<<(N * (N / 32) + 255) / 256, 256, 0, stream>>>(adj, bits, N, N * (N / 32));

    // 2) Wh = x @ W per head  (fp32 + bf16-transposed); wave = 16x64 strip
    {
        int waves = H * (N / 16);                       // 2048
        gemm_bf16_tn<4><<<waves / 4, 128, 0, stream>>>(xb, WtH, Whf, WhT, N, F, H);
    }

    // 3) s1/s2, softmax stats, fused P@Wh (+ELU) -> h_cat (wave = 16x64 strip)
    s_dots<<<(H * N + 255) / 256, 256, 0, stream>>>(Whf, a1h, a2h, s1h, s2h, D, N, H * N);
    softmax_stats<<<(H * N * 32 + 255) / 256, 256, 0, stream>>>(s1h, s2h, bits, mh, zh, N, N, H * N);
    attn_av<4><<<dim3(N / 16 / 8, H), 256, 0, stream>>>(s1h, s2h, mh, zh, bits, WhT,
                                                        N, nullptr, hcat, HD, /*elu=*/1);

    // 4) output layer: Wh_out = h_cat @ W_out; wave = 16x128 strip
    {
        int waves = N / 16;                             // 256
        gemm_bf16_tn<8><<<waves / 4, 128, 0, stream>>>(hcat, WtO, Wof, WoT, N, HD, 1);
    }
    s_dots<<<(N + 255) / 256, 256, 0, stream>>>(Wof, a1o, a2o, s1o, s2o, O, N, N);
    softmax_stats<<<(N * 32 + 255) / 256, 256, 0, stream>>>(s1o, s2o, bits, mo, zo, N, N, N);
    attn_av<8><<<dim3(N / 16 / 8, 1), 256, 0, stream>>>(s1o, s2o, mo, zo, bits, WoT,
                                                        N, out, nullptr, O, /*elu=*/0);
}
